// GeeArrYou_62405874811231
// MI455X (gfx1250) — matrix operand order repaired and checked by reading, not compiled, not run on hardware
//
#include <hip/hip_runtime.h>
#include <hip/hip_bf16.h>
#include <math.h>

#define BATCH 64
#define TLEN  2048
#define HID   256
#define G3    768          // 3*H, gate-concat dim
#define LAYERS 4
#define TC    128          // time-chunk so gx chunk (25MB) stays L2-resident
#define NCHUNK (TLEN / TC)
#define WELEM_PER_LAYER (48 * 8 * 32 * 16)   // [ntile][kstep][lane][i] halves
#define RTHREADS 512                         // 16 waves, 4/SIMD

typedef __attribute__((ext_vector_type(16))) _Float16 v16h;
typedef __attribute__((ext_vector_type(8)))  _Float16 v8h;
typedef __attribute__((ext_vector_type(8)))  float    v8f;

union AFrag { v16h v; v8h h[2]; _Float16 e[16]; };
union CFrag { v8f  v; float e[8]; };

#define LOG2E 1.4426950408889634f

// Hardware transcendentals (v_exp_f32 is exp2; ~1 ULP; TRANS pipe co-executes
// with the WMMA/XDL pipe). Saturation at +-inf is exact for both gates.
__device__ __forceinline__ float fast_sigmoid(float x) {
    float e = __builtin_amdgcn_exp2f(-LOG2E * x);
    return __builtin_amdgcn_rcpf(1.f + e);
}
__device__ __forceinline__ float fast_tanh(float x) {
    float e = __builtin_amdgcn_exp2f((2.f * LOG2E) * x);   // e^{2x}
    return 1.f - 2.f * __builtin_amdgcn_rcpf(e + 1.f);
}

// LDS address of hidden-state element (m, k) stored in WMMA A-fragment layout:
// a wave reads its A operand as 16 contiguous halves at [kstep*512 + lane*16].
// ISA A 16x32 f16 layout: element i of lane -> k = kstep*32 + (i>>3)*16 + (i&7) + (lane>>4)*8, m = lane&15.
__device__ __forceinline__ int fragaddr(int m, int k) {
    int ks   = k >> 5;
    int kl   = k & 31;
    int half = (kl >> 3) & 1;
    int lane = half * 16 + m;
    int i    = ((kl >> 4) << 3) + (kl & 7);
    return ks * 512 + lane * 16 + i;
}

// Issue this thread's share (6 x B128) of one 48KB gx step-slab: global -> LDS, async.
// Per wave that is 6 ASYNCcnt increments; completions are in-order per wave.
__device__ __forceinline__ void gx_async_copy(const float* __restrict__ gsrc,
                                              float* ldst, int tid) {
    unsigned lbase = (unsigned)(uintptr_t)ldst;   // low 32 bits of generic LDS ptr = LDS byte addr
#pragma unroll
    for (int q = 0; q < 6; ++q) {
        unsigned           loff  = lbase + (unsigned)((q * RTHREADS + tid) * 16);
        unsigned long long gaddr = (unsigned long long)(uintptr_t)gsrc
                                 + (unsigned long long)((q * RTHREADS + tid) * 16);
        asm volatile("global_load_async_to_lds_b128 %0, %1, off"
                     :: "v"(loff), "v"(gaddr) : "memory");
    }
}

// ---------------------------------------------------------------------------
// Shuffle Wi/Wh (f32 [L][768][256]) into f16 B-fragment layout:
// per layer: [ntile(48)][kstep(8)][lane(32)][i(16)]
// ISA B 32x16 f16 layout: lane holds column n = lane&15, k = (i>>3)*16 + (lane>>4)*8 + (i&7) (+ kstep*32).
// ---------------------------------------------------------------------------
__global__ void gru_wshuffle(const float* __restrict__ Wi, const float* __restrict__ Wh,
                             _Float16* __restrict__ wi16, _Float16* __restrict__ wh16) {
    int idx = blockIdx.x * 256 + threadIdx.x;
    const int per_mat = LAYERS * WELEM_PER_LAYER;
    if (idx >= 2 * per_mat) return;
    int which = idx / per_mat;
    int r     = idx - which * per_mat;
    int l     = r / WELEM_PER_LAYER;
    int r2    = r % WELEM_PER_LAYER;
    int ntile = r2 >> 12;          // /4096
    int r3    = r2 & 4095;
    int ks    = r3 >> 9;
    int r4    = r3 & 511;
    int lane  = r4 >> 4;
    int i     = r4 & 15;
    int j = ntile * 16 + (lane & 15);
    int k = ((i >> 3) * 16) + ((lane >> 4) * 8) + (i & 7) + ks * 32;
    const float* src = which ? Wh : Wi;
    float v = src[((long)l * G3 + j) * HID + k];
    (which ? wh16 : wi16)[r] = (_Float16)v;
}

// ---------------------------------------------------------------------------
// Input projection for one time-chunk: gx[t_loc][b][j] = act[b][t]@Wi^T
// Each wave: one 16-row M-tile (16 consecutive t for one b) x 3 N-tiles.
// 8192 waves = 1024 blocks x 8 waves. Templated on input dtype.
// ---------------------------------------------------------------------------
template <bool F32IN>
__global__ __launch_bounds__(256)
void gru_gx(const float* __restrict__ in_f32,       // layer 0 input
            const _Float16* __restrict__ in_f16,    // later-layer input
            const _Float16* __restrict__ wibuf,     // this layer's shuffled Wi
            float* __restrict__ gxc, int t0) {
    int tid    = threadIdx.x;
    int lane   = tid & 31;
    int waveId = blockIdx.x * 8 + (tid >> 5);   // 0..8191
    int g      = waveId & 15;                   // N-group: tiles g, g+16, g+32
    int mt     = waveId >> 4;                   // 0..511
    int b      = mt >> 3;
    int tbase  = t0 + (mt & 7) * 16;
    int m = lane & 15, half = lane >> 4;

    const _Float16* wb0 = wibuf + (long)((g     ) * 8) * 512;
    const _Float16* wb1 = wibuf + (long)((g + 16) * 8) * 512;
    const _Float16* wb2 = wibuf + (long)((g + 32) * 8) * 512;

    long rowbase = ((long)b * TLEN + tbase + m) * HID;
    v8f a0 = {}, a1 = {}, a2 = {};
#pragma unroll
    for (int ks = 0; ks < 8; ++ks) {
        AFrag af;
        int koff = ks * 32 + half * 8;
        if (F32IN) {
            const float* p = in_f32 + rowbase + koff;
#pragma unroll
            for (int q = 0; q < 8; ++q) {
                af.e[q]     = (_Float16)p[q];
                af.e[q + 8] = (_Float16)p[16 + q];
            }
        } else {
            const _Float16* p = in_f16 + rowbase + koff;
            af.h[0] = *(const v8h*)p;
            af.h[1] = *(const v8h*)(p + 16);
        }
        v16h b0 = *(const v16h*)(wb0 + (long)(ks * 32 + lane) * 16);
        v16h b1 = *(const v16h*)(wb1 + (long)(ks * 32 + lane) * 16);
        v16h b2 = *(const v16h*)(wb2 + (long)(ks * 32 + lane) * 16);
        a0 = __builtin_amdgcn_wmma_f32_16x16x32_f16(false, af.v, false, b0, (short)0, a0, false, false);
        a1 = __builtin_amdgcn_wmma_f32_16x16x32_f16(false, af.v, false, b1, (short)0, a1, false, false);
        a2 = __builtin_amdgcn_wmma_f32_16x16x32_f16(false, af.v, false, b2, (short)0, a2, false, false);
    }
    CFrag c0, c1, c2; c0.v = a0; c1.v = a1; c2.v = a2;
    int n = lane & 15, rb = (lane >> 4) * 8;
#pragma unroll
    for (int e = 0; e < 8; ++e) {
        int trow = (tbase - t0) + rb + e;
        float* dst = gxc + ((long)trow * BATCH + b) * G3;
        dst[(g     ) * 16 + n] = c0.e[e];
        dst[(g + 16) * 16 + n] = c1.e[e];
        dst[(g + 32) * 16 + n] = c2.e[e];
    }
}

// ---------------------------------------------------------------------------
// Recurrent scan for one time-chunk. 4 blocks x 512 threads (16 waves);
// block owns batch rows [16*blockIdx, +16). Wave w owns N-tiles {w, w+16, w+32}
// = gates {r, z, n} for hidden units j0 in [16w, 16w+16): its three WMMA
// accumulators hold r/z/n for the SAME elements, so gate math is done fully
// in registers (no gh LDS round-trip). Tiles w, w+16 keep B-frags in VGPRs
// (128 regs); tile w+32's B-frags are streamed from an LDS copy of that third
// of Wh. Hidden state: fp32 carry in registers (hreg), f16 A-fragment image
// in LDS for the cross-wave matmul broadcast. gx double-buffered in LDS via
// async global->LDS copies (ASYNCcnt pipelining).
// ---------------------------------------------------------------------------
__global__ __launch_bounds__(RTHREADS)
void gru_rec(const float* __restrict__ gxc,        // [TC][64][768], L2-resident
             const _Float16* __restrict__ whbuf,   // this layer's shuffled Wh
             const float* __restrict__ biL, const float* __restrict__ bhL,  // [768]
             float* __restrict__ h_state,          // [64][256] carry between chunks
             _Float16* __restrict__ act_out,       // [B][T][256] next layer input
             float* __restrict__ fin_out,          // d_out for last layer, else null
             int t0, int first_chunk) {
    __shared__ __align__(32) _Float16 hfrag[8 * 512];        // 8 KB, A-fragment layout
    __shared__ __align__(32) _Float16 whlds[16 * 8 * 512];   // 128 KB, Wh tiles 32..47
    __shared__ __align__(16) float gxbuf[2][16 * G3];        // 96 KB, async double buffer

    int tid    = threadIdx.x;
    int lane   = tid & 31;
    int wave   = tid >> 5;            // 0..15
    int b_base = blockIdx.x * 16;
    int n  = lane & 15;
    int rb = (lane >> 4) * 8;
    int j0 = wave * 16 + n;           // this lane's hidden unit (fixed all chunk)

    // per-lane bias scalars (fixed for the whole chunk)
    float bir  = biL[j0],           bhr = bhL[j0];
    float biz  = biL[HID + j0],     bhz = bhL[HID + j0];
    float bin_ = biL[2 * HID + j0], bhn = bhL[2 * HID + j0];

    // stage third tile-set of Wh into LDS (one-time, 128KB)
    {
        const v8h* src = (const v8h*)(whbuf + (long)32 * 8 * 512);
        v8h*       dst = (v8h*)whlds;
        for (int idx = tid; idx < 16 * 8 * 512 / 8; idx += RTHREADS) dst[idx] = src[idx];
    }

    // register-resident fp32 hidden state for this lane's 8 elements + hfrag image
    float hreg[8];
#pragma unroll
    for (int e = 0; e < 8; ++e) {
        int m = rb + e;
        float v = first_chunk ? 0.f : h_state[(b_base + m) * HID + j0];
        hreg[e] = v;
        hfrag[fragaddr(m, j0)] = (_Float16)v;
    }

    // VGPR-resident B-frags for tiles wave, wave+16
    v16h bf0[8], bf1[8];
#pragma unroll
    for (int ks = 0; ks < 8; ++ks) {
        bf0[ks] = *(const v16h*)(whbuf + (long)(( wave       * 8 + ks) * 32 + lane) * 16);
        bf1[ks] = *(const v16h*)(whbuf + (long)(((wave + 16) * 8 + ks) * 32 + lane) * 16);
    }
    // prologue: start async fill of step t0's gx slab
    gx_async_copy(gxc + (long)b_base * G3, &gxbuf[0][0], tid);
    __syncthreads();

    for (int t = t0; t < t0 + TC; ++t) {
        // ---- gh = h @ Wh^T on matrix cores ----
        v8f a0 = {}, a1 = {}, a2 = {};
#pragma unroll
        for (int ks = 0; ks < 8; ++ks) {
            v16h af = *(const v16h*)&hfrag[ks * 512 + lane * 16];
            v16h b2 = *(const v16h*)&whlds[(wave * 8 + ks) * 512 + lane * 16];
            a0 = __builtin_amdgcn_wmma_f32_16x16x32_f16(false, af, false, bf0[ks], (short)0, a0, false, false);
            a1 = __builtin_amdgcn_wmma_f32_16x16x32_f16(false, af, false, bf1[ks], (short)0, a1, false, false);
            a2 = __builtin_amdgcn_wmma_f32_16x16x32_f16(false, af, false, b2,      (short)0, a2, false, false);
        }
        // kick off next step's gx slab, then retire this step's (6 newest may
        // stay in flight; per-wave ASYNCcnt completes in order).
        if (t + 1 < t0 + TC) {
            gx_async_copy(gxc + ((long)(t + 1 - t0) * BATCH + b_base) * G3,
                          &gxbuf[(t + 1 - t0) & 1][0], tid);
            asm volatile("s_wait_asynccnt 0x6" ::: "memory");
        } else {
            asm volatile("s_wait_asynccnt 0x0" ::: "memory");
        }
        __syncthreads();   // all hfrag reads done; gx slab visible to all waves

        // ---- gate math fully in registers: c0=r, c1=z, c2=n for (m=rb+e, j0) ----
        CFrag c0, c1, c2; c0.v = a0; c1.v = a1; c2.v = a2;
        const float* gxr0 = &gxbuf[(t - t0) & 1][0];
#pragma unroll
        for (int e = 0; e < 8; ++e) {
            int m = rb + e;
            const float* gxr = gxr0 + m * G3;
            float xr = gxr[j0], xz = gxr[HID + j0], xn = gxr[2 * HID + j0];
            float r  = fast_sigmoid(xr + c0.e[e] + bir + bhr);
            float z  = fast_sigmoid(xz + c1.e[e] + biz + bhz);
            float nn = fast_tanh(xn + bin_ + r * (c2.e[e] + bhn));
            float hnew = (1.f - z) * nn + z * hreg[e];
            hreg[e] = hnew;
            hfrag[fragaddr(m, j0)] = (_Float16)hnew;
            long o = ((long)(b_base + m) * TLEN + t) * HID + j0;
            act_out[o] = (_Float16)hnew;
            if (fin_out) fin_out[o] = hnew;
        }
        __syncthreads();   // hfrag writes done before next step's matmul
    }
#pragma unroll
    for (int e = 0; e < 8; ++e)
        h_state[(b_base + rb + e) * HID + j0] = hreg[e];
}

// ---------------------------------------------------------------------------
extern "C" void kernel_launch(void* const* d_in, const int* in_sizes, int n_in,
                              void* d_out, int out_size, void* d_ws, size_t ws_size,
                              hipStream_t stream) {
    (void)in_sizes; (void)n_in; (void)out_size; (void)ws_size;
    const float* x  = (const float*)d_in[0];
    const float* Wi = (const float*)d_in[1];
    const float* Wh = (const float*)d_in[2];
    const float* bi = (const float*)d_in[3];
    const float* bh = (const float*)d_in[4];
    float* out = (float*)d_out;

    char* ws = (char*)d_ws;
    size_t off = 0;
    auto carve = [&](size_t bytes) -> void* {
        off = (off + 255) & ~(size_t)255;
        void* p = ws + off;
        off += bytes;
        return p;
    };
    _Float16* wi16  = (_Float16*)carve((size_t)LAYERS * WELEM_PER_LAYER * 2);
    _Float16* wh16  = (_Float16*)carve((size_t)LAYERS * WELEM_PER_LAYER * 2);
    _Float16* act0  = (_Float16*)carve((size_t)BATCH * TLEN * HID * 2);
    _Float16* act1  = (_Float16*)carve((size_t)BATCH * TLEN * HID * 2);
    float*    gxc   = (float*)   carve((size_t)TC * BATCH * G3 * 4);
    float*    hstat = (float*)   carve((size_t)BATCH * HID * 4);

    int shtot = 2 * LAYERS * WELEM_PER_LAYER;
    gru_wshuffle<<<(shtot + 255) / 256, 256, 0, stream>>>(Wi, Wh, wi16, wh16);

    for (int l = 0; l < LAYERS; ++l) {
        const _Float16* inf16 = (l == 0) ? nullptr : ((l & 1) ? act0 : act1);
        _Float16*       aout  = (l & 1) ? act1 : act0;
        for (int c = 0; c < NCHUNK; ++c) {
            int t0 = c * TC;
            if (l == 0)
                gru_gx<true><<<1024, 256, 0, stream>>>(x, nullptr,
                    wi16 + (size_t)l * WELEM_PER_LAYER, gxc, t0);
            else
                gru_gx<false><<<1024, 256, 0, stream>>>(nullptr, inf16,
                    wi16 + (size_t)l * WELEM_PER_LAYER, gxc, t0);
            gru_rec<<<4, RTHREADS, 0, stream>>>(gxc, wh16 + (size_t)l * WELEM_PER_LAYER,
                                                bi + l * G3, bh + l * G3, hstat, aout,
                                                (l == LAYERS - 1) ? out : nullptr, t0, c == 0);
        }
    }
}
